// MultiheadAttention_35527969472983
// MI455X (gfx1250) — matrix operand order repaired and checked by reading, not compiled, not run on hardware
//
#include <hip/hip_runtime.h>

// ---------------------------------------------------------------------------
// MI455X (gfx1250) multi-head attention forward.
//   B=2, S=2048, D=1024, H=16, dk=64, SCALING=1.0 (module default, no 1/sqrt).
// bf16 everywhere through v_wmma_f32_16x16x32_bf16 (fp32 accumulate).
// GEMMs: TDM (tensor_load_to_lds) double-buffered LDS staging of A/B tiles,
// one DMA per tile per block instead of per-wave duplicated global loads;
// sync via s_wait_tensorcnt + workgroup barrier (CDNA5 ISA).
// Attention: K stored transposed for contiguous score B-fragments, 3-pass
// softmax with score recompute (XDL recompute beats 536MB of probs traffic),
// head-mean of probs via no-return global_atomic_add_f32 into L2.
// Whole bf16 working set (~50MB) fits the 192MB L2. Workspace: 64MB of d_ws.
// ---------------------------------------------------------------------------

#define DEV static __device__ __forceinline__

typedef __attribute__((ext_vector_type(16))) __bf16 v16bf;
typedef __attribute__((ext_vector_type(8)))  __bf16 v8bf;
typedef __attribute__((ext_vector_type(8)))  float  v8f;
typedef __attribute__((ext_vector_type(4)))  unsigned int v4u;
typedef __attribute__((ext_vector_type(8)))  int v8i;
typedef __attribute__((ext_vector_type(4)))  int v4i;

constexpr int NB     = 2;
constexpr int SEQ    = 2048;
constexpr int DMODEL = 1024;
constexpr int NHEAD  = 16;
constexpr int HD     = 64;                 // head dim
constexpr int MTOT   = NB * SEQ;           // 4096 rows

DEV v16bf cat8(v8bf lo, v8bf hi) {
  return __builtin_shufflevector(lo, hi, 0,1,2,3,4,5,6,7,8,9,10,11,12,13,14,15);
}

// A-fragment (16x32, M x K, bf16): lane L holds row M=L&15; K chunks
// {kb..kb+7} and {kb+16..kb+23} with kb=(L>>4)*8 (ISA 7.12.2 16-bit A layout).
DEV v16bf load_a_frag(const __bf16* row, int kb) {
  v8bf lo = *(const v8bf*)(row + kb);
  v8bf hi = *(const v8bf*)(row + kb + 16);
  return cat8(lo, hi);
}

// B-fragment (32x16, K x N, bf16): lane L holds column N=L&15; K chunks
// {kb..kb+7} and {kb+16..kb+23} with kb=(L>>4)*8, row stride ld.
DEV v16bf load_b_frag(const __bf16* p, int ld, int kb) {
  v16bf v;
#pragma unroll
  for (int s = 0; s < 8; ++s) {
    v[s] = p[(size_t)(kb + s) * ld];
    v[8 + s] = p[(size_t)(kb + 16 + s) * ld];
  }
  return v;
}

DEV v8f wmma_bf16(v16bf a, v16bf b, v8f c) {
  return __builtin_amdgcn_wmma_f32_16x16x32_bf16(false, a, false, b,
                                                 (short)0, c, false, false);
}

DEV v8f vzero() {
  v8f z;
#pragma unroll
  for (int i = 0; i < 8; ++i) z[i] = 0.0f;
  return z;
}

// ---------------------------------------------------------------------------
// Tensor Data Mover: 2D tile (rows x cols, bf16) global -> LDS, packed rows.
// D# per CDNA5 ISA: group0 = count|lds|gaddr|type=2,
// group1 = mask0|data_size=2B|tensor dims|tile dims|dim0 stride. Groups 2/3
// zero (2D: tile_dim2=0 => unused). Uniform args -> SGPR descriptor.
// Toolchain here is the 6-arg builtin: (v4u, v8i, v4i, v4i, v8i, i32 cpol).
DEV void tdm_load_2d(unsigned int lds_off, const __bf16* gptr,
                     unsigned int tdim0, unsigned int tdim1,
                     unsigned int stride0,          // elements
                     unsigned int tile0, unsigned int tile1) {
  const unsigned long long ga = (unsigned long long)(uintptr_t)gptr;
  v4u g0;
  g0[0] = 1u;                                   // count=1, user descriptor
  g0[1] = lds_off;                              // lds_addr (bytes)
  g0[2] = (unsigned int)ga;                     // global_addr[31:0]
  g0[3] = ((unsigned int)(ga >> 32) & 0x01FFFFFFu) | (2u << 30);  // addr|type=2
  v8i g1;
  g1[0] = (int)(1u << 16);                      // data_size=1 -> 2 bytes
  g1[1] = (int)((tdim0 & 0xFFFFu) << 16);       // tensor_dim0[15:0]
  g1[2] = (int)((tdim0 >> 16) | ((tdim1 & 0xFFFFu) << 16));
  g1[3] = (int)((tdim1 >> 16) | (tile0 << 16)); // tile_dim0
  g1[4] = (int)tile1;                           // tile_dim1 (tile_dim2=0)
  g1[5] = (int)stride0;                         // tensor_dim0_stride[31:0]
  g1[6] = 0;                                    // stride0 hi | stride1 lo
  g1[7] = 0;
  const v4i z4 = {0, 0, 0, 0};
  const v8i z8 = {0, 0, 0, 0, 0, 0, 0, 0};
  __builtin_amdgcn_tensor_load_to_lds(g0, g1, z4, z4, z8, 0);
}

// ---------------------------------------------------------------------------
__global__ void zero_f32_kernel(float* __restrict__ p, int n) {
  int i = blockIdx.x * blockDim.x + threadIdx.x;
  if (i < n) p[i] = 0.0f;
}

// W [N,K] f32 -> WT [K,N] bf16 via 32x32 LDS tile (coalesced both sides).
__global__ void wt_transpose_kernel(const float* __restrict__ W,
                                    __bf16* __restrict__ WT) {
  __shared__ float t[32][33];
  const int tx = threadIdx.x, ty = threadIdx.y;   // blockDim (32,8)
  const int n0 = blockIdx.y * 32, k0 = blockIdx.x * 32;
#pragma unroll
  for (int i = 0; i < 32; i += 8)
    t[ty + i][tx] = W[(size_t)(n0 + ty + i) * DMODEL + k0 + tx];
  __syncthreads();
#pragma unroll
  for (int i = 0; i < 32; i += 8)
    WT[(size_t)(k0 + ty + i) * DMODEL + n0 + tx] = (__bf16)t[tx][ty + i];
}

__global__ void cvt_bf16_kernel(const float* __restrict__ x,
                                __bf16* __restrict__ y, int n) {
  int i = blockIdx.x * blockDim.x + threadIdx.x;
  if (i < n) y[i] = (__bf16)x[i];
}

// ---------------------------------------------------------------------------
// GEMM: out[m,n] = sum_k X[m,k]*WT[k,n] + bias[n].  X bf16 [4096,1024],
// WT bf16 [1024,1024]. Block: 256 thr (8 waves), tile 128x128; wave 64x32.
// TDM stages A (128x32) and B (32x128) tiles into double-buffered LDS; the
// DMA of step i+1 overlaps the 8 WMMAs of step i.
// MODE 0: bf16 store, head-split [B,H,S,hd]        (q, v projections)
// MODE 1: bf16 store, head-split transposed [B,H,hd,S]   (k projection)
// MODE 2: f32 store, row-major [M,D]               (output proj -> d_out)
template <int MODE>
__global__ void gemm_kernel(const __bf16* __restrict__ X,
                            const __bf16* __restrict__ WT,
                            const float* __restrict__ bias,
                            void* __restrict__ dst) {
  __shared__ __bf16 sA[2][128][32];   // 16KB: rows m-local, packed k
  __shared__ __bf16 sB[2][32][128];   // 16KB: rows k-local, packed n

  const int lane = threadIdx.x & 31;
  const int wave = threadIdx.x >> 5;          // 0..7
  const int mblk = blockIdx.x >> 3;           // 0..31
  const int nblk = blockIdx.x & 7;            // 0..7
  const int wmloc = (wave & 1) * 64;          // wave tile inside block
  const int wnloc = (wave >> 1) * 32;
  const int m0 = mblk * 128 + wmloc;
  const int n0 = nblk * 128 + wnloc;
  const int l15 = lane & 15, lhi = lane >> 4;

  v8f acc[4][2];
#pragma unroll
  for (int ms = 0; ms < 4; ++ms)
#pragma unroll
    for (int ns = 0; ns < 2; ++ns) acc[ms][ns] = vzero();

  const __bf16* aTile = X + (size_t)mblk * 128 * DMODEL;     // + k0
  const __bf16* bTile = WT + nblk * 128;                      // + k0*DMODEL

  const int steps = DMODEL / 32;   // 32
  if (wave == 0) {
    tdm_load_2d((unsigned int)(uintptr_t)&sA[0][0][0], aTile,
                DMODEL, MTOT, DMODEL, 32, 128);
    tdm_load_2d((unsigned int)(uintptr_t)&sB[0][0][0], bTile,
                DMODEL, DMODEL, DMODEL, 128, 32);
    __builtin_amdgcn_s_wait_tensorcnt(0);
  }
  __syncthreads();

  for (int i = 0; i < steps; ++i) {
    const int buf = i & 1;
    if (wave == 0 && (i + 1) < steps) {      // prefetch next tiles via TDM
      const int k1 = (i + 1) * 32;
      tdm_load_2d((unsigned int)(uintptr_t)&sA[buf ^ 1][0][0], aTile + k1,
                  DMODEL, MTOT, DMODEL, 32, 128);
      tdm_load_2d((unsigned int)(uintptr_t)&sB[buf ^ 1][0][0],
                  bTile + (size_t)k1 * DMODEL, DMODEL, DMODEL, DMODEL, 128, 32);
    }

    v16bf a[4], b[2];
#pragma unroll
    for (int ms = 0; ms < 4; ++ms)
      a[ms] = load_a_frag(&sA[buf][wmloc + ms * 16 + l15][0], lhi * 8);
#pragma unroll
    for (int ns = 0; ns < 2; ++ns)
      b[ns] = load_b_frag(&sB[buf][0][wnloc + ns * 16 + l15], 128, lhi * 8);
#pragma unroll
    for (int ms = 0; ms < 4; ++ms)
#pragma unroll
      for (int ns = 0; ns < 2; ++ns)
        acc[ms][ns] = wmma_bf16(a[ms], b[ns], acc[ms][ns]);

    if (wave == 0) __builtin_amdgcn_s_wait_tensorcnt(0);
    __syncthreads();
  }

#pragma unroll
  for (int ms = 0; ms < 4; ++ms) {
#pragma unroll
    for (int ns = 0; ns < 2; ++ns) {
      const int n  = n0 + ns * 16 + l15;
      const float bv = bias[n];
#pragma unroll
      for (int r = 0; r < 8; ++r) {                 // C/D: row r+8*lhi, col l15
        const int m = m0 + ms * 16 + r + 8 * lhi;
        const float val = acc[ms][ns][r] + bv;
        if (MODE == 2) {
          ((float*)dst)[(size_t)m * DMODEL + n] = val;
        } else {
          const int bb = m >> 11, s = m & (SEQ - 1);
          const int h  = n >> 6,  d = n & 63;
          if (MODE == 0)
            ((__bf16*)dst)[(((size_t)(bb * NHEAD + h) * SEQ) + s) * HD + d] =
                (__bf16)val;
          else
            ((__bf16*)dst)[(((size_t)(bb * NHEAD + h) * HD) + d) * SEQ + s] =
                (__bf16)val;
        }
      }
    }
  }
}

// ---------------------------------------------------------------------------
// Attention per (b, h, 64 q-rows): block = 4 waves, wave owns 16 q rows.
// Pass1 row-max, Pass2 row-sumexp, Pass3 probs -> mean atomics + attn@V.
__global__ void attn_kernel(const __bf16* __restrict__ qh,   // [B,H,S,hd]
                            const __bf16* __restrict__ kT,   // [B,H,hd,S]
                            const __bf16* __restrict__ vh,   // [B,H,S,hd]
                            __bf16* __restrict__ comb,       // [B,S,D] bf16
                            float* __restrict__ mean) {      // [B,S,S] f32
  __shared__ __bf16 lds_p[4][16][32];   // per-wave probs tile (C->A transpose)

  const int lane = threadIdx.x & 31, wave = threadIdx.x >> 5;   // 4 waves
  const int qt = blockIdx.x & 31;
  const int h  = (blockIdx.x >> 5) & 15;
  const int b  = blockIdx.x >> 9;
  const int bh = b * NHEAD + h;
  const int l15 = lane & 15, lhi = lane >> 4;

  const int sq = qt * 64 + wave * 16 + l15;       // this lane's A-layout q row
  const __bf16* qrow = qh + ((size_t)bh * SEQ + sq) * HD;
  const v16bf qa0 = load_a_frag(qrow, lhi * 8);        // dk 0..31
  const v16bf qa1 = load_a_frag(qrow + 32, lhi * 8);   // dk 32..63
  const __bf16* kb = kT + (size_t)bh * HD * SEQ;
  const __bf16* vb = vh + (size_t)bh * SEQ * HD;

  float mrow[8], lrow[8];
#pragma unroll
  for (int r = 0; r < 8; ++r) { mrow[r] = -3.0e38f; lrow[r] = 0.0f; }

  // ---- pass 1: row max (SCALING = 1.0, no scale applied) ----
  for (int kt = 0; kt < SEQ; kt += 16) {
    __builtin_prefetch(kb + (size_t)lane * SEQ + kt + 64, 0, 1);
    v8f c = vzero();
    c = wmma_bf16(qa0, load_b_frag(kb + kt + l15, SEQ, lhi * 8), c);
    c = wmma_bf16(qa1, load_b_frag(kb + (size_t)32 * SEQ + kt + l15, SEQ, lhi * 8), c);
#pragma unroll
    for (int r = 0; r < 8; ++r) mrow[r] = fmaxf(mrow[r], c[r]);
  }
#pragma unroll
  for (int r = 0; r < 8; ++r)
#pragma unroll
    for (int off = 1; off < 16; off <<= 1)
      mrow[r] = fmaxf(mrow[r], __shfl_xor(mrow[r], off, 32));

  // ---- pass 2: row sum of exp ----
  for (int kt = 0; kt < SEQ; kt += 16) {
    v8f c = vzero();
    c = wmma_bf16(qa0, load_b_frag(kb + kt + l15, SEQ, lhi * 8), c);
    c = wmma_bf16(qa1, load_b_frag(kb + (size_t)32 * SEQ + kt + l15, SEQ, lhi * 8), c);
#pragma unroll
    for (int r = 0; r < 8; ++r) lrow[r] += __expf(c[r] - mrow[r]);
  }
#pragma unroll
  for (int r = 0; r < 8; ++r)
#pragma unroll
    for (int off = 1; off < 16; off <<= 1)
      lrow[r] += __shfl_xor(lrow[r], off, 32);
  float rinv[8];
#pragma unroll
  for (int r = 0; r < 8; ++r) rinv[r] = 1.0f / lrow[r];

  // ---- pass 3: normalized probs -> head-mean atomics + attn @ V ----
  const float hscale = 1.0f / (float)NHEAD;
  v8f av[4];
#pragma unroll
  for (int ns = 0; ns < 4; ++ns) av[ns] = vzero();

  for (int kt = 0; kt < SEQ; kt += 32) {
    v8f c0 = vzero(), c1 = vzero();
    c0 = wmma_bf16(qa0, load_b_frag(kb + kt + l15, SEQ, lhi * 8), c0);
    c0 = wmma_bf16(qa1, load_b_frag(kb + (size_t)32 * SEQ + kt + l15, SEQ, lhi * 8), c0);
    c1 = wmma_bf16(qa0, load_b_frag(kb + kt + 16 + l15, SEQ, lhi * 8), c1);
    c1 = wmma_bf16(qa1, load_b_frag(kb + (size_t)32 * SEQ + kt + 16 + l15, SEQ, lhi * 8), c1);

#pragma unroll
    for (int r = 0; r < 8; ++r) {
      const int row = r + 8 * lhi;                 // 0..15 in wave tile
      const int gq  = qt * 64 + wave * 16 + row;   // global q row
      const float p0 = __expf(c0[r] - mrow[r]) * rinv[r];
      const float p1 = __expf(c1[r] - mrow[r]) * rinv[r];
      lds_p[wave][row][l15]      = (__bf16)p0;
      lds_p[wave][row][16 + l15] = (__bf16)p1;
      float* mp = mean + ((size_t)b * SEQ + gq) * SEQ + kt;
      atomicAdd(mp + l15,      p0 * hscale);       // global_atomic_add_f32
      atomicAdd(mp + 16 + l15, p1 * hscale);
    }
    // same-wave DS store -> DS load by other lanes: drain DScnt; LDS ops of a
    // wave stay in order, so no workgroup barrier needed.
    asm volatile("s_wait_dscnt 0" ::: "memory");

    const v16bf pa = load_a_frag(&lds_p[wave][l15][0], lhi * 8);
#pragma unroll
    for (int ns = 0; ns < 4; ++ns) {
      v16bf bfv = load_b_frag(vb + (size_t)kt * HD + ns * 16 + l15, HD, lhi * 8);
      av[ns] = wmma_bf16(pa, bfv, av[ns]);
    }
  }

  // store combined [B,S,H*hd] bf16 for the output projection
#pragma unroll
  for (int ns = 0; ns < 4; ++ns)
#pragma unroll
    for (int r = 0; r < 8; ++r) {
      const int row = r + 8 * lhi;
      const int gq  = qt * 64 + wave * 16 + row;
      comb[((size_t)b * SEQ + gq) * DMODEL + h * HD + ns * 16 + l15] =
          (__bf16)av[ns][r];
    }
}

// ---------------------------------------------------------------------------
extern "C" void kernel_launch(void* const* d_in, const int* in_sizes, int n_in,
                              void* d_out, int out_size, void* d_ws,
                              size_t ws_size, hipStream_t stream) {
  const float* Q  = (const float*)d_in[0];
  const float* Kx = (const float*)d_in[1];
  const float* V  = (const float*)d_in[2];
  const float* Wq = (const float*)d_in[3];
  const float* bq = (const float*)d_in[4];
  const float* Wk = (const float*)d_in[5];
  const float* bk = (const float*)d_in[6];
  const float* Wv = (const float*)d_in[7];
  const float* bv = (const float*)d_in[8];
  const float* Wo = (const float*)d_in[9];
  const float* bo = (const float*)d_in[10];

  char* ws = (char*)d_ws;
  const size_t MB = 1u << 20;
  // workspace map (64MB total)
  __bf16* wtq  = (__bf16*)(ws + 0 * MB);    // Wq^T  bf16 [K,N]  2MB
  __bf16* wtk  = (__bf16*)(ws + 2 * MB);
  __bf16* wtv  = (__bf16*)(ws + 4 * MB);
  __bf16* wto  = (__bf16*)(ws + 6 * MB);
  __bf16* xq   = (__bf16*)(ws + 8 * MB);    // bf16 activations  8MB each
  __bf16* xk   = (__bf16*)(ws + 16 * MB);
  __bf16* xv   = (__bf16*)(ws + 24 * MB);
  __bf16* qhp  = (__bf16*)(ws + 32 * MB);   // q  [B,H,S,hd]     8MB
  __bf16* ktp  = (__bf16*)(ws + 40 * MB);   // k  [B,H,hd,S]     8MB
  __bf16* vhp  = (__bf16*)(ws + 48 * MB);   // v  [B,H,S,hd]     8MB
  __bf16* comb = (__bf16*)(ws + 56 * MB);   // attn combined     8MB

  float* out  = (float*)d_out;                       // [B,S,D] f32
  float* mean = out + (size_t)MTOT * DMODEL;         // [B,S,S] f32

  // 1. zero the probs-mean accumulator (atomically accumulated below)
  {
    const int n = NB * SEQ * SEQ;
    zero_f32_kernel<<<(n + 255) / 256, 256, 0, stream>>>(mean, n);
  }
  // 2. weights -> transposed bf16
  {
    dim3 tb(32, 8), tg(DMODEL / 32, DMODEL / 32);
    wt_transpose_kernel<<<tg, tb, 0, stream>>>(Wq, wtq);
    wt_transpose_kernel<<<tg, tb, 0, stream>>>(Wk, wtk);
    wt_transpose_kernel<<<tg, tb, 0, stream>>>(Wv, wtv);
    wt_transpose_kernel<<<tg, tb, 0, stream>>>(Wo, wto);
  }
  // 3. activations -> bf16
  {
    const int n = MTOT * DMODEL;
    const int g = (n + 255) / 256;
    cvt_bf16_kernel<<<g, 256, 0, stream>>>(Q, xq, n);
    cvt_bf16_kernel<<<g, 256, 0, stream>>>(Kx, xk, n);
    cvt_bf16_kernel<<<g, 256, 0, stream>>>(V, xv, n);
  }
  // 4. projections (TDM + WMMA GEMMs); K stored transposed for score GEMM
  gemm_kernel<0><<<256, 256, 0, stream>>>(xq, wtq, bq, qhp);
  gemm_kernel<1><<<256, 256, 0, stream>>>(xk, wtk, bk, ktp);
  gemm_kernel<0><<<256, 256, 0, stream>>>(xv, wtv, bv, vhp);
  // 5. attention (WMMA scores + softmax + WMMA attn@V + mean-probs atomics)
  attn_kernel<<<NB * NHEAD * (SEQ / 64), 128, 0, stream>>>(qhp, ktp, vhp, comb,
                                                           mean);
  // 6. output projection -> f32 d_out
  gemm_kernel<2><<<256, 256, 0, stream>>>(comb, wto, bo, out);
}
